// LinearizedContext_12326556139663
// MI455X (gfx1250) — compile-verified, matrix operand order checked by reading, not run on hardware
//
#include <hip/hip_runtime.h>

typedef _Float16 v8hf __attribute__((ext_vector_type(8)));
typedef _Float16 v16h __attribute__((ext_vector_type(16)));
typedef float    v8f  __attribute__((ext_vector_type(8)));

#define NN   4096
#define HH   256
#define CC   151
#define KBIG 416   // 256 (H) + 160 (C padded), = 13 * 32

// ------------------------------------------------------------------
// small prep kernels
// ------------------------------------------------------------------

__global__ void k_f16_copy(const float* __restrict__ src, _Float16* __restrict__ dst, int n) {
  int i = blockIdx.x * blockDim.x + threadIdx.x;
  if (i < n) dst[i] = (_Float16)src[i];
}

// Wt[h][k] = (f16) W[k][h]   (256x256)
__global__ void k_transpose16(const float* __restrict__ W, _Float16* __restrict__ Wt) {
  int idx = blockIdx.x * blockDim.x + threadIdx.x;
  int h = idx >> 8, k = idx & 255;
  Wt[h * HH + k] = (_Float16)W[k * HH + h];
}

// label_scores[c] = sum_b Co[c][b]
__global__ void k_label(const float* __restrict__ Co, float* __restrict__ ls) {
  int c = threadIdx.x;
  if (c < CC) {
    float a = 0.f;
    for (int b = 0; b < CC; ++b) a += Co[c * CC + b];
    ls[c] = a;
  }
}

// s[i] = D[i,:] . ls
__global__ void k_s(const float* __restrict__ D, const float* __restrict__ ls,
                    float* __restrict__ s) {
  int i = blockIdx.x * blockDim.x + threadIdx.x;
  float a = 0.f;
  for (int c = 0; c < CC; ++c) a += D[(size_t)i * CC + c] * ls[c];
  s[i] = a;
}

// U[j,c] = sum_a D[j,a] * Co[a,c]  -> Bbig[j, 256+c] (f16), pad 151..159 with 0
__global__ void k_U(const float* __restrict__ D, const float* __restrict__ Co,
                    _Float16* __restrict__ Bbig) {
  __shared__ float Drow[CC];
  int j = blockIdx.x;
  int t = threadIdx.x;               // 160 threads
  if (t < CC) Drow[t] = D[(size_t)j * CC + t];
  __syncthreads();
  if (t < CC) {
    float a = 0.f;
    for (int aa = 0; aa < CC; ++aa) a += Drow[aa] * Co[aa * CC + t];  // coalesced over t
    Bbig[(size_t)j * KBIG + HH + t] = (_Float16)a;
  } else {
    Bbig[(size_t)j * KBIG + HH + t] = (_Float16)0.f;
  }
}

// Abig[i, 256+c] = (f16)(w_ps * D[i,c]), pad tail with 0
__global__ void k_packD(const float* __restrict__ D, const float* __restrict__ wsc,
                        _Float16* __restrict__ Abig) {
  int i = blockIdx.x, t = threadIdx.x;  // 160 threads
  float w_ps = wsc[3 * HH];
  if (t < CC) Abig[(size_t)i * KBIG + HH + t] = (_Float16)(w_ps * D[(size_t)i * CC + t]);
  else        Abig[(size_t)i * KBIG + HH + t] = (_Float16)0.f;
}

// rowAdd[i] = obj[i,:].w_o + w_os*s[i] ; colAdd[j] = sub[j,:].w_s + w_ss*s[j] + b
__global__ void k_vec(const _Float16* __restrict__ Bbig, const _Float16* __restrict__ obj16,
                      const float* __restrict__ s, const float* __restrict__ wsc,
                      const float* __restrict__ bsc,
                      float* __restrict__ rowAdd, float* __restrict__ colAdd) {
  int i = blockIdx.x * blockDim.x + threadIdx.x;
  float sv = 0.f, ov = 0.f;
  for (int h = 0; h < HH; ++h) {
    sv += (float)Bbig[(size_t)i * KBIG + h] * wsc[HH + h];
    ov += (float)obj16[(size_t)i * HH + h] * wsc[2 * HH + h];
  }
  float w_ss = wsc[3 * HH + 1], w_os = wsc[3 * HH + 2];
  rowAdd[i] = ov + w_os * s[i];
  colAdd[i] = sv + w_ss * s[i] + bsc[0];
}

// ------------------------------------------------------------------
// WMMA GEMM core: C(128x256 per WG) = A(M x K) * B(N x K)^T, f16 in, f32 acc
// 8 waves: wave grid 2x4, each wave 64x64 = 4x4 tiles of 16x16.
// Per K-step per wave: 16 WMMAs fed by 8 fragments (16 ds_load_b128)
// -> 1.0 LDS b128 loads per WMMA (balanced vs 64-bank LDS bandwidth).
// ------------------------------------------------------------------

union F16x16 { v16h v; v8hf h[2]; };

template <int KTOT>
__device__ __forceinline__ void gemm_core(const _Float16* __restrict__ A,
                                          const _Float16* __restrict__ B,
                                          int bm, int bn, v8f acc[4][4]) {
  __shared__ _Float16 As[128 * 32];   // 8 KB
  __shared__ _Float16 Bs[256 * 32];   // 16 KB

  const int tid   = threadIdx.x;
  const int lane  = tid & 31;
  const int wave  = tid >> 5;
  const int wm    = wave >> 2;   // 0..1  (64-row slab)
  const int wn    = wave & 3;    // 0..3  (64-col slab)
  const int hf    = lane >> 4;   // K-half select per ISA 16-bit A layout
  const int lm    = lane & 15;
  const int ldrow = tid >> 2;    // 0..63
  const int ldseg = tid & 3;     // 0..3 (8 f16 each)

  for (int k0 = 0; k0 < KTOT; k0 += 32) {
    v8hf a0 = *(const v8hf*)(A + (size_t)(bm + ldrow)       * KTOT + k0 + ldseg * 8);
    v8hf a1 = *(const v8hf*)(A + (size_t)(bm + 64 + ldrow)  * KTOT + k0 + ldseg * 8);
    v8hf b0 = *(const v8hf*)(B + (size_t)(bn + ldrow)       * KTOT + k0 + ldseg * 8);
    v8hf b1 = *(const v8hf*)(B + (size_t)(bn + 64 + ldrow)  * KTOT + k0 + ldseg * 8);
    v8hf b2 = *(const v8hf*)(B + (size_t)(bn + 128 + ldrow) * KTOT + k0 + ldseg * 8);
    v8hf b3 = *(const v8hf*)(B + (size_t)(bn + 192 + ldrow) * KTOT + k0 + ldseg * 8);
    __syncthreads();   // previous iteration's LDS reads complete
    *(v8hf*)(As + ldrow * 32 + ldseg * 8)         = a0;
    *(v8hf*)(As + (64 + ldrow) * 32 + ldseg * 8)  = a1;
    *(v8hf*)(Bs + ldrow * 32 + ldseg * 8)         = b0;
    *(v8hf*)(Bs + (64 + ldrow) * 32 + ldseg * 8)  = b1;
    *(v8hf*)(Bs + (128 + ldrow) * 32 + ldseg * 8) = b2;
    *(v8hf*)(Bs + (192 + ldrow) * 32 + ldseg * 8) = b3;
    __syncthreads();

    F16x16 af[4], bf[4];
#pragma unroll
    for (int tm = 0; tm < 4; ++tm) {
      int r = wm * 64 + tm * 16 + lm;
      af[tm].h[0] = *(const v8hf*)(As + r * 32 + hf * 8);        // K = hf*8 .. +7
      af[tm].h[1] = *(const v8hf*)(As + r * 32 + hf * 8 + 16);   // K = 16+hf*8 .. +7
    }
#pragma unroll
    for (int tn = 0; tn < 4; ++tn) {
      int c = wn * 64 + tn * 16 + lm;
      bf[tn].h[0] = *(const v8hf*)(Bs + c * 32 + hf * 8);
      bf[tn].h[1] = *(const v8hf*)(Bs + c * 32 + hf * 8 + 16);
    }
#pragma unroll
    for (int tm = 0; tm < 4; ++tm)
#pragma unroll
      for (int tn = 0; tn < 4; ++tn)
        acc[tm][tn] = __builtin_amdgcn_wmma_f32_16x16x32_f16(
            false, af[tm].v, false, bf[tn].v, (short)0, acc[tm][tn], false, false);
  }
}

// MLP GEMM: dst1[row, col] = (f16)(relu(acc + bias[col]) * scale?[col]) ; dst2 = raw relu
__global__ __launch_bounds__(256) void k_gemm_mlp(const _Float16* __restrict__ A,
                                                  const _Float16* __restrict__ Wt,
                                                  const float* __restrict__ bias,
                                                  const float* __restrict__ scale,
                                                  _Float16* __restrict__ dst1, int ld1,
                                                  _Float16* __restrict__ dst2) {
  int bm = blockIdx.x * 128, bn = 0;     // N = 256 fits one block column
  v8f acc[4][4] = {};
  gemm_core<HH>(A, Wt, bm, bn, acc);

  const int lane = threadIdx.x & 31, wave = threadIdx.x >> 5;
  const int wm = wave >> 2, wn = wave & 3, hf = lane >> 4, lm = lane & 15;
#pragma unroll
  for (int tm = 0; tm < 4; ++tm)
#pragma unroll
    for (int tn = 0; tn < 4; ++tn) {
      int row0 = bm + wm * 64 + tm * 16 + hf * 8;
      int col  = bn + wn * 64 + tn * 16 + lm;
      float b  = bias[col];
      float sc = scale ? scale[col] : 1.0f;
#pragma unroll
      for (int r = 0; r < 8; ++r) {
        int row = row0 + r;
        float v = acc[tm][tn][r] + b;
        v = v > 0.f ? v : 0.f;
        dst1[(size_t)row * ld1 + col] = (_Float16)(v * sc);
        if (dst2) dst2[(size_t)row * HH + col] = (_Float16)v;
      }
    }
}

// Main fused GEMM: out[i*N+j] = sigmoid(Abig[i,:].Bbig[j,:] + rowAdd[i] + colAdd[j])
__global__ __launch_bounds__(256) void k_gemm_main(const _Float16* __restrict__ A,
                                                   const _Float16* __restrict__ B,
                                                   const float* __restrict__ rowAdd,
                                                   const float* __restrict__ colAdd,
                                                   float* __restrict__ out) {
  int bm = blockIdx.x * 128, bn = blockIdx.y * 256;
  v8f acc[4][4] = {};
  gemm_core<KBIG>(A, B, bm, bn, acc);

  const int lane = threadIdx.x & 31, wave = threadIdx.x >> 5;
  const int wm = wave >> 2, wn = wave & 3, hf = lane >> 4, lm = lane & 15;
#pragma unroll
  for (int tm = 0; tm < 4; ++tm)
#pragma unroll
    for (int tn = 0; tn < 4; ++tn) {
      int row0 = bm + wm * 64 + tm * 16 + hf * 8;
      int col  = bn + wn * 64 + tn * 16 + lm;
      float ca = colAdd[col];
#pragma unroll
      for (int r = 0; r < 8; ++r) {
        int row = row0 + r;
        float v = acc[tm][tn][r] + rowAdd[row] + ca;
        out[(size_t)row * NN + col] = 1.0f / (1.0f + __expf(-v));
      }
    }
}

// ------------------------------------------------------------------
// launcher
// ------------------------------------------------------------------
extern "C" void kernel_launch(void* const* d_in, const int* in_sizes, int n_in,
                              void* d_out, int out_size, void* d_ws, size_t ws_size,
                              hipStream_t stream) {
  const float* prepro = (const float*)d_in[0];   // (N,H)
  const float* D      = (const float*)d_in[1];   // (N,C)
  const float* Co     = (const float*)d_in[2];   // (C,C)
  const float* W_sub  = (const float*)d_in[3];   // (H,H)
  const float* b_sub  = (const float*)d_in[4];   // (H,)
  const float* W_obj  = (const float*)d_in[5];   // (H,H)
  const float* b_obj  = (const float*)d_in[6];   // (H,)
  const float* wsc    = (const float*)d_in[7];   // (3H+3,)
  const float* bsc    = (const float*)d_in[8];   // scalar
  float* out = (float*)d_out;

  char* w = (char*)d_ws;
  size_t off = 0;
  auto carve = [&](size_t bytes) { char* p = w + off; off += (bytes + 255) & ~(size_t)255; return p; };
  _Float16* Af16  = (_Float16*)carve((size_t)NN * HH * 2);   // 2 MB
  _Float16* WsT   = (_Float16*)carve((size_t)HH * HH * 2);   // 128 KB
  _Float16* WoT   = (_Float16*)carve((size_t)HH * HH * 2);   // 128 KB
  _Float16* Abig  = (_Float16*)carve((size_t)NN * KBIG * 2); // 3.25 MB
  _Float16* Bbig  = (_Float16*)carve((size_t)NN * KBIG * 2); // 3.25 MB
  _Float16* obj16 = (_Float16*)carve((size_t)NN * HH * 2);   // 2 MB
  float* ls     = (float*)carve(CC * 4);
  float* s      = (float*)carve(NN * 4);
  float* rowAdd = (float*)carve(NN * 4);
  float* colAdd = (float*)carve(NN * 4);
  (void)ws_size; (void)n_in; (void)in_sizes; (void)out_size;

  // prep
  k_f16_copy<<<(NN * HH) / 256, 256, 0, stream>>>(prepro, Af16, NN * HH);
  k_transpose16<<<(HH * HH) / 256, 256, 0, stream>>>(W_sub, WsT);
  k_transpose16<<<(HH * HH) / 256, 256, 0, stream>>>(W_obj, WoT);
  k_label<<<1, 256, 0, stream>>>(Co, ls);
  k_s<<<NN / 256, 256, 0, stream>>>(D, ls, s);
  k_U<<<NN, 160, 0, stream>>>(D, Co, Bbig);          // Bbig cols 256..415
  k_packD<<<NN, 160, 0, stream>>>(D, wsc, Abig);     // Abig cols 256..415

  // MLP GEMMs (fill Abig/Bbig cols 0..255)
  k_gemm_mlp<<<dim3(NN / 128, 1), 256, 0, stream>>>(
      Af16, WsT, b_sub, nullptr, Bbig, KBIG, nullptr);             // sub
  k_gemm_mlp<<<dim3(NN / 128, 1), 256, 0, stream>>>(
      Af16, WoT, b_obj, wsc /*w_mul = wsc[0:256]*/, Abig, KBIG, obj16);  // obj*w_mul

  // epilogue vectors
  k_vec<<<NN / 256, 256, 0, stream>>>(Bbig, obj16, s, wsc, bsc, rowAdd, colAdd);

  // fused big GEMM + sigmoid
  k_gemm_main<<<dim3(NN / 128, NN / 256), 256, 0, stream>>>(Abig, Bbig, rowAdd, colAdd, out);
}
